// QLinear_43138651521149
// MI455X (gfx1250) — matrix operand order repaired and checked by reading, not compiled, not run on hardware
//
#include <hip/hip_runtime.h>
#include <stdint.h>

// ---------- types ----------
typedef __bf16 v16bf __attribute__((ext_vector_type(16)));
typedef __bf16 v8bf  __attribute__((ext_vector_type(8)));
typedef __bf16 v4bf  __attribute__((ext_vector_type(4)));
typedef float  v8f   __attribute__((ext_vector_type(8)));

#define M_DIM 8192
#define N_DIM 4096
#define K_DIM 4096

#define KSTEP 64                 // K elements staged per pipeline stage
#define LDS_STRIDE 72            // 64 + 8 pad elems (144B row) -> distinct 16B bank groups

// power-of-two log quantization (matches reference expquantize, n=2)
__device__ __forceinline__ float expquantize(float x) {
    float c = fminf(fmaxf(x, -1.0f), 1.0f);
    float a = fabsf(c);
    // log2f(0) = -inf -> rintf -> -inf -> exp2f -> 0
    float v = exp2f(rintf(log2f(a)));
    v = (v < 0.25f) ? 0.0f : v;
    return copysignf(v, c);
}

// ---------- preprocessing kernels ----------
__global__ void cvt_x_bf16(const float* __restrict__ in, __bf16* __restrict__ out, int n) {
    int i = (blockIdx.x * blockDim.x + threadIdx.x) * 4;
    if (i >= n) return;
    float4 v = *(const float4*)(in + i);
    v4bf q;
    q.x = (__bf16)v.x; q.y = (__bf16)v.y; q.z = (__bf16)v.z; q.w = (__bf16)v.w;
    *(v4bf*)(out + i) = q;
}

__global__ void quant_w_bf16(const float* __restrict__ in, __bf16* __restrict__ out, int n) {
    int i = (blockIdx.x * blockDim.x + threadIdx.x) * 4;
    if (i >= n) return;
    float4 v = *(const float4*)(in + i);
    v4bf q;
    q.x = (__bf16)expquantize(v.x);
    q.y = (__bf16)expquantize(v.y);
    q.z = (__bf16)expquantize(v.z);
    q.w = (__bf16)expquantize(v.w);
    *(v4bf*)(out + i) = q;
}

__global__ void quant_b_f32(const float* __restrict__ in, float* __restrict__ out, int n) {
    int i = blockIdx.x * blockDim.x + threadIdx.x;
    if (i < n) out[i] = expquantize(in[i]);
}

// ---------- staging: one 16-byte chunk global -> LDS (async, ASYNCcnt) ----------
// GLOBAL_LOAD_ASYNC_TO_LDS_B128: VDST = 32-bit LDS byte address, VADDR = 64-bit
// global address (GV mode). Generic pointers to __shared__ carry the LDS offset
// in their low 32 bits.
__device__ __forceinline__ void copy16_g2lds(const __bf16* g, __bf16* l) {
    uint32_t lds_addr = (uint32_t)(uintptr_t)l;
    asm volatile("global_load_async_to_lds_b128 %0, %1, off"
                 :
                 : "v"(lds_addr), "v"(g)
                 : "memory");
}

__device__ __forceinline__ void async_wait_all() {
    asm volatile("s_wait_asynccnt 0x0" ::: "memory");
}

// Stage A(128 x 64) and B(128 x 64) bf16 tiles into LDS.
// 1024 16B chunks per tile; 256 threads x 4 chunks each.
__device__ __forceinline__ void stage_tiles(
    const __bf16* __restrict__ gA, const __bf16* __restrict__ gB, int k0,
    __bf16* sA, __bf16* sB, int t)
{
#pragma unroll
    for (int i = 0; i < 4; ++i) {
        const int c   = t + i * 256;
        const int row = c >> 3;            // 8 chunks of 8 elems per 64-elem row
        const int col = (c & 7) * 8;
        copy16_g2lds(gA + (size_t)row * K_DIM + k0 + col, sA + row * LDS_STRIDE + col);
    }
#pragma unroll
    for (int i = 0; i < 4; ++i) {
        const int c   = t + i * 256;
        const int row = c >> 3;
        const int col = (c & 7) * 8;
        copy16_g2lds(gB + (size_t)row * K_DIM + k0 + col, sB + row * LDS_STRIDE + col);
    }
}

__device__ __forceinline__ v16bf frag_from_lds(const __bf16* base, int off0, int off1) {
    v8bf lo = *(const v8bf*)(base + off0);
    v8bf hi = *(const v8bf*)(base + off1);
    return __builtin_shufflevector(lo, hi,
        0, 1, 2, 3, 4, 5, 6, 7, 8, 9, 10, 11, 12, 13, 14, 15);
}

// ---------- WMMA GEMM with double-buffered async LDS pipeline ----------
// out[M,N] = xb[M,K] @ wb[N,K]^T + bq[N]
// 256 threads = 8 waves (2M x 4N); WG tile 128x128; wave tile 64x32 (4x2 accums).
__global__ __launch_bounds__(256) void qlinear_wmma(
    const __bf16* __restrict__ xb, const __bf16* __restrict__ wb,
    const float* __restrict__ bq, float* __restrict__ out)
{
    __shared__ __bf16 sA[2][128 * LDS_STRIDE];
    __shared__ __bf16 sB[2][128 * LDS_STRIDE];

    const int t    = threadIdx.x;
    const int lane = t & 31;                 // wave32
    const int wave = t >> 5;
    const int wm   = wave & 1;               // 0..1 (M)
    const int wn   = wave >> 1;              // 0..3 (N)
    const int l15  = lane & 15;
    const int hi   = lane >> 4;

    const int tileN = blockIdx.x * 128;
    const int tileM = blockIdx.y * 128;
    const __bf16* gA = xb + (size_t)tileM * K_DIM;
    const __bf16* gB = wb + (size_t)tileN * K_DIM;

    // per-wave fragment row bases inside the LDS tiles
    int aRow[4], bRow[2];
#pragma unroll
    for (int mi = 0; mi < 4; ++mi) aRow[mi] = (wm * 64 + mi * 16 + l15) * LDS_STRIDE;
#pragma unroll
    for (int ni = 0; ni < 2; ++ni) bRow[ni] = (wn * 32 + ni * 16 + l15) * LDS_STRIDE;

    const v8f vzero = {0.f, 0.f, 0.f, 0.f, 0.f, 0.f, 0.f, 0.f};
    v8f acc[4][2];
#pragma unroll
    for (int mi = 0; mi < 4; ++mi)
#pragma unroll
        for (int ni = 0; ni < 2; ++ni)
            acc[mi][ni] = vzero;

    const int nsteps = K_DIM / KSTEP;        // 64 stages
    stage_tiles(gA, gB, 0, sA[0], sB[0], t);

    for (int ks = 0; ks < nsteps; ++ks) {
        const int cur = ks & 1;
        async_wait_all();                    // own async copies for buffer `cur` done
        __syncthreads();                     // staged tile visible; prev readers done
        if (ks + 1 < nsteps)
            stage_tiles(gA, gB, (ks + 1) * KSTEP, sA[1 - cur], sB[1 - cur], t);

        // consume the 64 staged K in two 16x16x32 WMMA sub-steps
#pragma unroll
        for (int ksub = 0; ksub < KSTEP; ksub += 32) {
            // B: lanes 0-15 K k..k+15, lanes 16-31 K k+16..k+31 (contiguous per lane)
            v16bf b0 = frag_from_lds(sB[cur] + bRow[0], ksub + hi * 16, ksub + hi * 16 + 8);
            v16bf b1 = frag_from_lds(sB[cur] + bRow[1], ksub + hi * 16, ksub + hi * 16 + 8);
#pragma unroll
            for (int mi = 0; mi < 4; ++mi) {
                // A: lanes 0-15 K k..k+15, lanes 16-31 K k+16..k+31 (contiguous per lane)
                v16bf a = frag_from_lds(sA[cur] + aRow[mi],
                                        ksub + hi * 16, ksub + hi * 16 + 8);
                acc[mi][0] = __builtin_amdgcn_wmma_f32_16x16x32_bf16(
                    false, a, false, b0, (short)0, acc[mi][0], false, false);
                acc[mi][1] = __builtin_amdgcn_wmma_f32_16x16x32_bf16(
                    false, a, false, b1, (short)0, acc[mi][1], false, false);
            }
        }
    }

    // Epilogue: C/D layout -> VGPR j: M = (lane>=16 ? 8 : 0) + j, N = lane&15
#pragma unroll
    for (int ni = 0; ni < 2; ++ni) {
        const int col = tileN + wn * 32 + ni * 16 + l15;
        const float bb = bq[col];
#pragma unroll
        for (int mi = 0; mi < 4; ++mi) {
            const int row0 = tileM + wm * 64 + mi * 16 + hi * 8;
            float* op = out + (size_t)row0 * N_DIM + col;
#pragma unroll
            for (int j = 0; j < 8; ++j) {
                op[(size_t)j * N_DIM] = acc[mi][ni][j] + bb;
            }
        }
    }
}

extern "C" void kernel_launch(void* const* d_in, const int* in_sizes, int n_in,
                              void* d_out, int out_size, void* d_ws, size_t ws_size,
                              hipStream_t stream) {
    const float* x    = (const float*)d_in[0];   // [8192, 4096] f32
    const float* w    = (const float*)d_in[1];   // [4096, 4096] f32
    const float* bias = (const float*)d_in[2];   // [4096] f32
    float* out = (float*)d_out;                  // [8192, 4096] f32

    // workspace layout: x bf16 | w bf16 | bias f32
    char* ws = (char*)d_ws;
    __bf16* xb = (__bf16*)ws;
    __bf16* wb = (__bf16*)(ws + (size_t)M_DIM * K_DIM * sizeof(__bf16));
    float*  bq = (float*)(ws + (size_t)M_DIM * K_DIM * sizeof(__bf16)
                             + (size_t)N_DIM * K_DIM * sizeof(__bf16));

    const int nx = M_DIM * K_DIM;
    const int nw = N_DIM * K_DIM;
    cvt_x_bf16<<<nx / 4 / 256, 256, 0, stream>>>(x, xb, nx);
    quant_w_bf16<<<nw / 4 / 256, 256, 0, stream>>>(w, wb, nw);
    quant_b_f32<<<N_DIM / 256, 256, 0, stream>>>(bias, bq, N_DIM);

    dim3 grid(N_DIM / 128, M_DIM / 128);
    qlinear_wmma<<<grid, 256, 0, stream>>>(xb, wb, bq, out);
}
